// SoftmaxFreeAttentionKernel_2250562863659
// MI455X (gfx1250) — matrix operand order repaired and checked by reading, not compiled, not run on hardware
//
#include <hip/hip_runtime.h>
#include <math.h>

#define BB 32
#define HH 8
#define NN 3136
#define DD 64
#define MM 49
#define HWW 56
#define BH 256
#define GY 7
#define TOK_PER_GY 448
#define CHUNKS 14

typedef __attribute__((ext_vector_type(16))) __bf16 v16bf;
typedef __attribute__((ext_vector_type(8)))  float  v8f;

__device__ __forceinline__ __bf16 f2bf(float f) {
  union { float f; unsigned u; } a; a.f = f;
  unsigned r = a.u + 0x7FFFu + ((a.u >> 16) & 1u);   // round-to-nearest-even
  union { unsigned short s; __bf16 b; } o; o.s = (unsigned short)(r >> 16);
  return o.b;
}

__device__ __forceinline__ v8f v8zero() {
  v8f r;
#pragma unroll
  for (int i = 0; i < 8; ++i) r[i] = 0.f;
  return r;
}

__device__ __forceinline__ v8f wmma_bf16(v16bf a, v16bf b, v8f c) {
  return __builtin_amdgcn_wmma_f32_16x16x32_bf16(false, a, false, b, (short)0, c, false, false);
}

// A-matrix (16x32 bf16): lane holds row r=lane&15; VGPR v pair K = 2*(v&3)+((v>>2)<<4)+hi*8
__device__ __forceinline__ v16bf lds_tile_a(const __bf16* p, int sr, int sk, int lane) {
  int r = lane & 15, hi = (lane >> 4) & 1;
  v16bf a;
#pragma unroll
  for (int v = 0; v < 8; ++v) {
    int k = 2 * (v & 3) + ((v >> 2) << 4) + (hi << 3);
    a[2 * v]     = p[r * sr + k * sk];
    a[2 * v + 1] = p[r * sr + (k + 1) * sk];
  }
  return a;
}

// B-matrix (32x16 bf16): lane holds col n=lane&15; VGPR v pair K = 2*(v&3)+((v>>2)<<4)+hi*8
__device__ __forceinline__ v16bf lds_tile_b(const __bf16* p, int sk, int sn, int lane) {
  int n = lane & 15, hi = (lane >> 4) & 1;
  v16bf b;
#pragma unroll
  for (int v = 0; v < 8; ++v) {
    int k = 2 * (v & 3) + ((v >> 2) << 4) + (hi << 3);
    b[2 * v]     = p[k * sk + n * sn];
    b[2 * v + 1] = p[(k + 1) * sk + n * sn];
  }
  return b;
}

// ---------------- Kernel 0: init scalars ----------------
__global__ void kinit(unsigned* maxrs) {
  if (threadIdx.x == 0 && blockIdx.x == 0) *maxrs = 0u;
}

// ---------------- Kernel 1: landmarks (patch GEMM + LN + GELU) ----------------
__global__ __launch_bounds__(256)
void kland(const float* __restrict__ Q, const float* __restrict__ W,
           const float* __restrict__ ln_g, const float* __restrict__ ln_b,
           float* __restrict__ ql, float* __restrict__ qlsq) {
  int bh = blockIdx.x, tid = threadIdx.x;
  __shared__ float patch[64][64];   // [t = kh*8+kw][din]
  __shared__ float red[64][4];
  __shared__ float qlr[49][64];
  const float scale = 0.3535533906f;  // 64^-0.25
  const float* Qb = Q + (size_t)bh * NN * DD;
  for (int m = 0; m < MM; ++m) {
    int my = m / 7, mx = m % 7;
    for (int i = tid; i < 64 * 64; i += 256) {
      int t = i >> 6, din = i & 63;
      int kh = t >> 3, kw = t & 7;
      int tok = (my * 8 + kh) * HWW + (mx * 8 + kw);
      patch[t][din] = Qb[(size_t)tok * DD + din] * scale;
    }
    __syncthreads();
    int dout = tid >> 2, part = tid & 3;
    float acc = 0.f;
    for (int din = part * 16; din < part * 16 + 16; ++din) {
      const float* wp = W + ((size_t)(dout * 64 + din)) * 64;
#pragma unroll 8
      for (int t = 0; t < 64; ++t) acc += patch[t][din] * wp[t];
    }
    red[dout][part] = acc;
    __syncthreads();
    if (tid < 64) qlr[m][tid] = red[tid][0] + red[tid][1] + red[tid][2] + red[tid][3];
    __syncthreads();
  }
  if (tid < MM) {
    float mu = 0.f;
    for (int d = 0; d < 64; ++d) mu += qlr[tid][d];
    mu *= (1.f / 64.f);
    float var = 0.f;
    for (int d = 0; d < 64; ++d) { float t = qlr[tid][d] - mu; var += t * t; }
    var *= (1.f / 64.f);
    float rs = rsqrtf(var + 1e-5f);
    float ss = 0.f;
    for (int d = 0; d < 64; ++d) {
      float x = (qlr[tid][d] - mu) * rs * ln_g[d] + ln_b[d];
      float g = 0.5f * x * (1.f + erff(x * 0.70710678f));   // exact GELU
      ql[((size_t)bh * MM + tid) * DD + d] = g;
      ss += g * g;
    }
    qlsq[bh * 64 + tid] = ss;
  }
  if (tid >= MM && tid < 64) qlsq[bh * 64 + tid] = 0.f;
}

// ---------------- Kernel 2: k2, rowsum max, mat2 = k2@k2 ----------------
__global__ __launch_bounds__(256)
void kk2(const float* __restrict__ ql, const float* __restrict__ qlsq,
         float* __restrict__ k2, float* __restrict__ mat2, unsigned* __restrict__ maxrs) {
  int bh = blockIdx.x, tid = threadIdx.x;
  __shared__ float Qls[49][64];
  __shared__ float k2s[49][49];
  __shared__ float rs[49];
  for (int i = tid; i < 49 * 64; i += 256) Qls[i / 64][i & 63] = ql[(size_t)bh * MM * DD + i];
  __syncthreads();
  for (int i = tid; i < 49 * 49; i += 256) {
    int r = i / 49, c = i % 49;
    float dot = 0.f;
    for (int d = 0; d < 64; ++d) dot += Qls[r][d] * Qls[c][d];
    float v = expf(-0.5f * (qlsq[bh * 64 + r] + qlsq[bh * 64 + c] - 2.f * dot));
    k2s[r][c] = v;
    k2[(size_t)bh * 2401 + i] = v;
  }
  __syncthreads();
  if (tid < 49) { float s = 0.f; for (int c = 0; c < 49; ++c) s += k2s[tid][c]; rs[tid] = s; }
  __syncthreads();
  if (tid == 0) {
    float mx = 0.f;
    for (int r = 0; r < 49; ++r) mx = fmaxf(mx, rs[r]);
    atomicMax(maxrs, __float_as_uint(mx));   // valid: positive floats
  }
  for (int i = tid; i < 49 * 49; i += 256) {
    int r = i / 49, c = i % 49;
    float dot = 0.f;
    for (int k = 0; k < 49; ++k) dot += k2s[r][k] * k2s[k][c];
    mat2[(size_t)bh * 2401 + i] = dot;
  }
}

// ---------------- Kernel 3: global alpha-halving loop ----------------
__global__ __launch_bounds__(256)
void kalpha(const float* __restrict__ mat2, const unsigned* __restrict__ maxrs,
            float* __restrict__ alpha_out) {
  __shared__ float red[256];
  int tid = threadIdx.x;
  float mx = __uint_as_float(*maxrs);
  float alpha = 2.f / (mx * mx);
  for (int it = 0; it < 10; ++it) {
    float lmax = 0.f;
    for (int idx = tid; idx < BH * MM; idx += 256) {
      int bh = idx / MM, j = idx % MM;
      const float* mp = mat2 + (size_t)bh * 2401 + j;
      float cs = 0.f;
      for (int i = 0; i < MM; ++i) {
        float v = ((i == j) ? 1.f : 0.f) - alpha * mp[i * 49];
        cs += fabsf(v);
      }
      lmax = fmaxf(lmax, cs);
    }
    red[tid] = lmax;
    __syncthreads();
    for (int s = 128; s > 0; s >>= 1) {
      if (tid < s) red[tid] = fmaxf(red[tid], red[tid + s]);
      __syncthreads();
    }
    float pn = red[0];
    __syncthreads();
    if (pn <= 1.01f) break;   // uniform across block
    alpha *= 0.5f;
  }
  if (tid == 0) *alpha_out = alpha;
}

// ---------------- Kernel 4: Newton inversion (20 iters, LDS) ----------------
__global__ __launch_bounds__(256)
void knewton(const float* __restrict__ k2, const float* __restrict__ alpha_p,
             float* __restrict__ k2inv) {
  int bh = blockIdx.x, tid = threadIdx.x;
  __shared__ float matS[2401], bufA[2401], bufB[2401], bufT[2401];
  float alpha = *alpha_p;
  for (int i = tid; i < 2401; i += 256) { float v = k2[(size_t)bh * 2401 + i]; matS[i] = v; bufA[i] = alpha * v; }
  __syncthreads();
  float* Vm = bufA; float* Um = bufB;
  for (int it = 0; it < 20; ++it) {
    for (int i = tid; i < 2401; i += 256) {
      int r = i / 49, c = i % 49; float s = 0.f;
      for (int k = 0; k < 49; ++k) s += matS[r * 49 + k] * Vm[k * 49 + c];
      bufT[i] = s;
    }
    __syncthreads();
    for (int i = tid; i < 2401; i += 256) {
      int r = i / 49, c = i % 49; float s = 0.f;
      for (int k = 0; k < 49; ++k) s += Vm[r * 49 + k] * bufT[k * 49 + c];
      Um[i] = 2.f * Vm[i] - s;
    }
    __syncthreads();
    float* t = Vm; Vm = Um; Um = t;
  }
  for (int i = tid; i < 2401; i += 256) k2inv[(size_t)bh * 2401 + i] = Vm[i];
}

// ---------------- Kernel 5 (WMMA): kv = k1^T @ V, fused score recompute ----------------
__global__ __launch_bounds__(128)
void kkv(const float* __restrict__ Q, const float* __restrict__ Vin,
         const float* __restrict__ ql, const float* __restrict__ qlsq_g,
         float* __restrict__ kvp) {
  int bh = blockIdx.x, gy = blockIdx.y;
  int tid = threadIdx.x, lane = tid & 31, wave = tid >> 5;
  const float scale = 0.3535533906f;
  __shared__ __align__(16) char smem[58112];
  __bf16* Qlb  = (__bf16*)smem;                     // [64][64] (m, d), rows>=49 zero
  float*  qlsq = (float*)(smem + 8192);             // [64]
  float*  qssq = (float*)(smem + 8448);             // [4][32]
  __bf16* Qsb  = (__bf16*)(smem + 8960);            // [4][32][64] (token, d)
  __bf16* Vb   = (__bf16*)(smem + 8960 + 16384);    // [4][32][64]
  char*   over = smem + 8960 + 32768;
  __bf16* Sb   = (__bf16*)over;                     // [4][64][32] (m, token)   during loop
  float*  kvred = (float*)over;                     // [64][64]                 after loop

  for (int i = tid; i < 4096; i += 128) {
    int r = i >> 6;
    Qlb[i] = (r < MM) ? f2bf(ql[(size_t)bh * MM * DD + i]) : f2bf(0.f);
  }
  for (int i = tid; i < 64; i += 128) qlsq[i] = qlsq_g[bh * 64 + i];
  __syncthreads();

  v16bf a_ql[4][2];
#pragma unroll
  for (int mt = 0; mt < 4; ++mt)
#pragma unroll
    for (int kh = 0; kh < 2; ++kh)
      a_ql[mt][kh] = lds_tile_a(Qlb + (mt * 16) * 64 + kh * 32, 64, 1, lane);

  v8f acc[4][4];
#pragma unroll
  for (int mt = 0; mt < 4; ++mt)
#pragma unroll
    for (int dt = 0; dt < 4; ++dt) acc[mt][dt] = v8zero();

  int hi = (lane >> 4) & 1, nl_ = lane & 15;
  for (int c = wave; c < CHUNKS; c += 4) {
    int n0 = gy * TOK_PER_GY + c * 32;
    // stage this wave's 32-token chunk (Q scaled + V) as bf16, row sq-norms
    {
      const float* qs = Q + ((size_t)bh * NN + n0 + lane) * DD;
      const float* vs = Vin + ((size_t)bh * NN + n0 + lane) * DD;
      __bf16* qd = Qsb + wave * 2048 + lane * 64;
      __bf16* vd = Vb + wave * 2048 + lane * 64;
      float ss = 0.f;
      for (int d = 0; d < 64; ++d) {
        float v = qs[d] * scale; ss += v * v; qd[d] = f2bf(v);
        vd[d] = f2bf(vs[d]);
      }
      qssq[wave * 32 + lane] = ss;
    }
    // stage 1: S^T[64m, 32tok] = Ql_pad @ Qs^T  (K = 64, 2 steps)
    v16bf b_qs[2][2];
#pragma unroll
    for (int kh = 0; kh < 2; ++kh)
#pragma unroll
      for (int nt = 0; nt < 2; ++nt)
        b_qs[kh][nt] = lds_tile_b(Qsb + wave * 2048 + (nt * 16) * 64 + kh * 32, 1, 64, lane);
    v8f S[4][2];
#pragma unroll
    for (int mt = 0; mt < 4; ++mt)
#pragma unroll
      for (int nt = 0; nt < 2; ++nt) {
        v8f t = wmma_bf16(a_ql[mt][0], b_qs[0][nt], v8zero());
        S[mt][nt] = wmma_bf16(a_ql[mt][1], b_qs[1][nt], t);
      }
    // transform: exp(-(||q||^2 + ||l||^2 - 2 q.l)/2), restripe to A-layout via LDS
    __bf16* sbp = Sb + wave * 2048;
#pragma unroll
    for (int mt = 0; mt < 4; ++mt)
#pragma unroll
      for (int nt = 0; nt < 2; ++nt)
#pragma unroll
        for (int r = 0; r < 8; ++r) {
          int m = mt * 16 + r + hi * 8;
          int nl = nt * 16 + nl_;
          float val = expf(-0.5f * (qssq[wave * 32 + nl] + qlsq[m] - 2.f * S[mt][nt][r]));
          sbp[m * 32 + nl] = f2bf(val);
        }
    // stage 2: kv += S^T @ V  (K = 32)
    v16bf b_v[4];
#pragma unroll
    for (int dt = 0; dt < 4; ++dt)
      b_v[dt] = lds_tile_b(Vb + wave * 2048 + dt * 16, 64, 1, lane);
#pragma unroll
    for (int mt = 0; mt < 4; ++mt) {
      v16bf a_s = lds_tile_a(sbp + mt * 16 * 32, 32, 1, lane);
#pragma unroll
      for (int dt = 0; dt < 4; ++dt)
        acc[mt][dt] = wmma_bf16(a_s, b_v[dt], acc[mt][dt]);
    }
  }
  __syncthreads();   // Sb region now free -> kvred
  for (int w = 0; w < 4; ++w) {
    if (wave == w) {
#pragma unroll
      for (int mt = 0; mt < 4; ++mt)
#pragma unroll
        for (int dt = 0; dt < 4; ++dt)
#pragma unroll
          for (int r = 0; r < 8; ++r) {
            int m = mt * 16 + r + hi * 8;
            int d = dt * 16 + nl_;
            if (w == 0) kvred[m * 64 + d] = acc[mt][dt][r];
            else        kvred[m * 64 + d] += acc[mt][dt][r];
          }
    }
    __syncthreads();
  }
  for (int i = tid; i < 4096; i += 128)
    kvp[((size_t)(bh * GY + gy)) * 4096 + i] = kvred[i];
}

// ---------------- Kernel 6: Y = k2inv @ (sum partials), zero-padded ----------------
__global__ __launch_bounds__(256)
void kyy(const float* __restrict__ kvp, const float* __restrict__ k2inv,
         float* __restrict__ Y) {
  int bh = blockIdx.x, tid = threadIdx.x;
  __shared__ float k2i[2401];
  __shared__ float kvs[49 * 64];
  for (int i = tid; i < 2401; i += 256) k2i[i] = k2inv[(size_t)bh * 2401 + i];
  for (int i = tid; i < 49 * 64; i += 256) {
    float s = 0.f;
    for (int g = 0; g < GY; ++g) s += kvp[((size_t)(bh * GY + g)) * 4096 + i];
    kvs[i] = s;
  }
  __syncthreads();
  for (int i = tid; i < 4096; i += 256) {
    int m = i >> 6, d = i & 63;
    float s = 0.f;
    if (m < MM) for (int k = 0; k < MM; ++k) s += k2i[m * 49 + k] * kvs[k * 64 + d];
    Y[(size_t)bh * 4096 + i] = s;   // rows 49..63 zero -> annihilate garbage S cols
  }
}

// ---------------- Kernel 7 (WMMA): X = k1 @ Y, fused score recompute ----------------
__global__ __launch_bounds__(128)
void kx(const float* __restrict__ Q, const float* __restrict__ ql,
        const float* __restrict__ qlsq_g, const float* __restrict__ Y,
        float* __restrict__ out) {
  int bh = blockIdx.x, gy = blockIdx.y;
  int tid = threadIdx.x, lane = tid & 31, wave = tid >> 5;
  const float scale = 0.3535533906f;
  __shared__ __align__(16) char smem[49920];
  __bf16* Qlb  = (__bf16*)smem;                  // [64][64] (m,d), rows>=49 zero
  __bf16* Yb   = (__bf16*)(smem + 8192);         // [64][64] (m,d), rows>=49 zero
  float*  qlsq = (float*)(smem + 16384);         // [64]
  float*  qssq = (float*)(smem + 16640);         // [4][32]
  __bf16* Qsb  = (__bf16*)(smem + 17152);        // [4][32][64] (token,d)
  __bf16* Sb   = (__bf16*)(smem + 17152 + 16384);// [4][32][64] (token,m)

  for (int i = tid; i < 4096; i += 128) {
    int r = i >> 6;
    Qlb[i] = (r < MM) ? f2bf(ql[(size_t)bh * MM * DD + i]) : f2bf(0.f);
    Yb[i] = f2bf(Y[(size_t)bh * 4096 + i]);
  }
  for (int i = tid; i < 64; i += 128) qlsq[i] = qlsq_g[bh * 64 + i];
  __syncthreads();

  v16bf b_ql[2][4];  // (k=d, n=m) from Qlb[m][d]
#pragma unroll
  for (int kh = 0; kh < 2; ++kh)
#pragma unroll
    for (int mt = 0; mt < 4; ++mt)
      b_ql[kh][mt] = lds_tile_b(Qlb + (mt * 16) * 64 + kh * 32, 1, 64, lane);
  v16bf b_y[2][4];   // (k=m, n=d) from Yb[m][d]
#pragma unroll
  for (int kh = 0; kh < 2; ++kh)
#pragma unroll
    for (int dt = 0; dt < 4; ++dt)
      b_y[kh][dt] = lds_tile_b(Yb + (kh * 32) * 64 + dt * 16, 64, 1, lane);

  int hi = (lane >> 4) & 1, nl_ = lane & 15;
  for (int c = wave; c < CHUNKS; c += 4) {
    int n0 = gy * TOK_PER_GY + c * 32;
    {
      const float* qs = Q + ((size_t)bh * NN + n0 + lane) * DD;
      __bf16* qd = Qsb + wave * 2048 + lane * 64;
      float ss = 0.f;
      for (int d = 0; d < 64; ++d) { float v = qs[d] * scale; ss += v * v; qd[d] = f2bf(v); }
      qssq[wave * 32 + lane] = ss;
    }
    // S[32tok, 64m] = Qs @ Ql^T
    v8f S[2][4];
#pragma unroll
    for (int rt = 0; rt < 2; ++rt) {
      v16bf a0 = lds_tile_a(Qsb + wave * 2048 + rt * 16 * 64, 64, 1, lane);
      v16bf a1 = lds_tile_a(Qsb + wave * 2048 + rt * 16 * 64 + 32, 64, 1, lane);
#pragma unroll
      for (int mt = 0; mt < 4; ++mt) {
        v8f t = wmma_bf16(a0, b_ql[0][mt], v8zero());
        S[rt][mt] = wmma_bf16(a1, b_ql[1][mt], t);
      }
    }
    // transform + restripe [token][m]
    __bf16* sbp = Sb + wave * 2048;
#pragma unroll
    for (int rt = 0; rt < 2; ++rt)
#pragma unroll
      for (int mt = 0; mt < 4; ++mt)
#pragma unroll
        for (int r = 0; r < 8; ++r) {
          int tl = rt * 16 + r + hi * 8;
          int m = mt * 16 + nl_;
          float val = expf(-0.5f * (qssq[wave * 32 + tl] + qlsq[m] - 2.f * S[rt][mt][r]));
          sbp[tl * 64 + m] = f2bf(val);
        }
    // X = S @ Y_pad, stream to output
#pragma unroll
    for (int rt = 0; rt < 2; ++rt) {
      v16bf a0 = lds_tile_a(sbp + rt * 16 * 64, 64, 1, lane);
      v16bf a1 = lds_tile_a(sbp + rt * 16 * 64 + 32, 64, 1, lane);
#pragma unroll
      for (int dt = 0; dt < 4; ++dt) {
        v8f X = wmma_bf16(a0, b_y[0][dt], v8zero());
        X = wmma_bf16(a1, b_y[1][dt], X);
#pragma unroll
        for (int r = 0; r < 8; ++r) {
          int tok = n0 + rt * 16 + r + hi * 8;
          int d = dt * 16 + nl_;
          out[((size_t)bh * NN + tok) * DD + d] = X[r];
        }
      }
    }
  }
}

// ---------------- Kernel 8: depthwise 3x3 conv over heads, += into out ----------------
__global__ __launch_bounds__(256)
void kconv(const float* __restrict__ Vin, const float* __restrict__ w,
           float* __restrict__ out) {
  int blk = blockIdx.x;         // bh*56 + y
  int bh = blk / HWW, y = blk % HWW;
  int h = bh % HH;
  float wv[9];
#pragma unroll
  for (int i = 0; i < 9; ++i) wv[i] = w[h * 9 + i];
  const float* Vb = Vin + (size_t)bh * NN * DD;
  float* ob = out + ((size_t)bh * NN + (size_t)y * HWW) * DD;
  for (int i = threadIdx.x; i < HWW * DD; i += 256) {
    int x = i / DD, d = i % DD;
    float acc = 0.f;
#pragma unroll
    for (int dy = 0; dy < 3; ++dy) {
      int yy = y + dy - 1;
      if (yy < 0 || yy >= HWW) continue;
#pragma unroll
      for (int dx = 0; dx < 3; ++dx) {
        int xx = x + dx - 1;
        if (xx < 0 || xx >= HWW) continue;
        acc += Vb[((size_t)yy * HWW + xx) * DD + d] * wv[dy * 3 + dx];
      }
    }
    ob[i] += acc;
  }
}

extern "C" void kernel_launch(void* const* d_in, const int* in_sizes, int n_in,
                              void* d_out, int out_size, void* d_ws, size_t ws_size,
                              hipStream_t stream) {
  (void)in_sizes; (void)n_in; (void)out_size; (void)ws_size;
  const float* Q   = (const float*)d_in[0];
  const float* V   = (const float*)d_in[1];
  const float* qw  = (const float*)d_in[2];
  const float* lng = (const float*)d_in[3];
  const float* lnb = (const float*)d_in[4];
  const float* cw  = (const float*)d_in[5];
  float* outp = (float*)d_out;

  float* ws    = (float*)d_ws;
  float* ql    = ws;                  // 256*49*64   = 802816
  float* qlsq  = ql + 802816;         // 256*64      = 16384
  float* k2    = qlsq + 16384;        // 256*2401    = 614656
  float* mat2  = k2 + 614656;
  float* k2inv = mat2 + 614656;
  float* kvp   = k2inv + 614656;      // 256*7*4096  = 7340032
  float* Y     = kvp + 7340032;       // 256*4096    = 1048576
  unsigned* maxrs = (unsigned*)(Y + 1048576);
  float* alpha = (float*)(maxrs + 1);

  kinit<<<1, 32, 0, stream>>>(maxrs);
  kland<<<BH, 256, 0, stream>>>(Q, qw, lng, lnb, ql, qlsq);
  kk2<<<BH, 256, 0, stream>>>(ql, qlsq, k2, mat2, maxrs);
  kalpha<<<1, 256, 0, stream>>>(mat2, maxrs, alpha);
  knewton<<<BH, 256, 0, stream>>>(k2, alpha, k2inv);
  kkv<<<dim3(BH, GY), 128, 0, stream>>>(Q, V, ql, qlsq, kvp);
  kyy<<<BH, 256, 0, stream>>>(kvp, k2inv, Y);
  kx<<<dim3(BH, GY), 128, 0, stream>>>(Q, ql, qlsq, Y, outp);
  kconv<<<BH * HWW, 256, 0, stream>>>(V, cw, outp);
}
